// GMNLayer_27994596835762
// MI455X (gfx1250) — matrix operand order repaired and checked by reading, not compiled, not run on hardware
//
#include <hip/hip_runtime.h>
#include <hip/hip_bf16.h>

typedef __attribute__((ext_vector_type(16))) _Float16 v16h;
typedef __attribute__((ext_vector_type(8)))  _Float16 v8h;
typedef __attribute__((ext_vector_type(4)))  _Float16 v4h;
typedef __attribute__((ext_vector_type(8)))  float    v8f;

#define NWI 4             // waves per block, isolated kernel
#define TBI (NWI * 32)
#define NWS 2             // waves per block, stick kernel (bigger LDS/wave)
#define TBS (NWS * 32)

// ---------------------------------------------------------------------------
// Fences / async helpers
// ---------------------------------------------------------------------------
__device__ __forceinline__ void lds_fence() {
  __builtin_amdgcn_wave_barrier();
  asm volatile("s_wait_dscnt 0" ::: "memory");
  __builtin_amdgcn_wave_barrier();
}

template <int N>
__device__ __forceinline__ void wait_async() {
  __builtin_amdgcn_wave_barrier();
  asm volatile("s_wait_asynccnt %0" :: "i"(N) : "memory");
  __builtin_amdgcn_wave_barrier();
}

// Per-lane async DMA: 16 bytes global -> LDS, tracked by ASYNCcnt (no VGPR
// staging; CDNA5 GLOBAL_LOAD_ASYNC_TO_LDS_B128, GV addressing mode).
__device__ __forceinline__ void async_gather_b128(const float* gp, void* lp) {
  asm volatile("global_load_async_to_lds_b128 %0, %1, off"
               :: "v"((unsigned int)(uintptr_t)lp), "v"(gp)
               : "memory");
}

// Issue one 16x64-f32 tile gather (indexed rows) as 8 async B128 ops/lane.
__device__ __forceinline__ void gather_tile_async(float* bufF, const float* feat,
                                                  const int* idx, int base, int cnt,
                                                  int lane) {
#pragma unroll
  for (int u = 0; u < 8; ++u) {
    const int e = lane + u * 32;       // 0..255 float4 units
    const int row = e >> 4;
    const int c4 = e & 15;
    int s = base + row; if (s > cnt - 1) s = cnt - 1;
    const int g = idx[s];
    async_gather_b128(feat + (size_t)g * 64 + c4 * 4, bufF + row * 64 + c4 * 4);
  }
}

// ---------------------------------------------------------------------------
// WMMA fragment loaders (CDNA5 16x16x32 f16 layouts, wave32).
//
// A (16x32): lanes 0-15 row m=lane, K = kb+{0..7, 16..23}; lanes 16-31 the
// +8 halves. B (32x16): weights transposed (out-major) in LDS; lane's column
// n = lane&15, K = kb+{0..7, 16..23} + (lane>>4)*8 (same order as A).
// C/D: element v -> row v + 8*(lane>>4), col lane&15.
// ---------------------------------------------------------------------------
__device__ __forceinline__ v16h load_a_frag(const _Float16* buf, int lane, int kb) {
  const int hs = lane >> 4, m = lane & 15;
  const _Float16* p = buf + m * 64 + kb + hs * 8;
  v8h lo = *(const v8h*)(p);
  v8h hi = *(const v8h*)(p + 16);
  return __builtin_shufflevector(lo, hi, 0, 1, 2, 3, 4, 5, 6, 7,
                                 8, 9, 10, 11, 12, 13, 14, 15);
}

// A-fragment from an f32 LDS tile (async-gathered), converting to f16 here.
__device__ __forceinline__ v16h load_a_frag_f32(const float* buf, int lane, int kb) {
  const int hs = lane >> 4, m = lane & 15;
  const float* p = buf + m * 64 + kb + hs * 8;
  const float4 f0 = *(const float4*)(p);
  const float4 f1 = *(const float4*)(p + 4);
  const float4 f2 = *(const float4*)(p + 16);
  const float4 f3 = *(const float4*)(p + 20);
  v16h a;
  a[0]  = (_Float16)f0.x; a[1]  = (_Float16)f0.y; a[2]  = (_Float16)f0.z; a[3]  = (_Float16)f0.w;
  a[4]  = (_Float16)f1.x; a[5]  = (_Float16)f1.y; a[6]  = (_Float16)f1.z; a[7]  = (_Float16)f1.w;
  a[8]  = (_Float16)f2.x; a[9]  = (_Float16)f2.y; a[10] = (_Float16)f2.z; a[11] = (_Float16)f2.w;
  a[12] = (_Float16)f3.x; a[13] = (_Float16)f3.y; a[14] = (_Float16)f3.z; a[15] = (_Float16)f3.w;
  return a;
}

__device__ __forceinline__ v16h load_b_frag(const _Float16* wt, int lane, int nb, int kb) {
  const int hs = lane >> 4, n = lane & 15;
  const _Float16* p = wt + (nb * 16 + n) * 64 + kb + hs * 8;
  v8h lo = *(const v8h*)(p);
  v8h hi = *(const v8h*)(p + 16);
  return __builtin_shufflevector(lo, hi, 0, 1, 2, 3, 4, 5, 6, 7,
                                 8, 9, 10, 11, 12, 13, 14, 15);
}

// out = act(A @ W + b) for a 16-row tile; A supplied as two prebuilt frags.
// 4 N-blocks x 2 K-steps = 8 v_wmma_f32_16x16x32_f16.
__device__ __forceinline__ void layer64_core(v16h a0, v16h a1, _Float16* out,
                                             const _Float16* Wt, const float* bias,
                                             int lane, bool relu) {
  const int hs = lane >> 4, n = lane & 15;
#pragma unroll
  for (int nb = 0; nb < 4; ++nb) {
    const float bv = bias[nb * 16 + n];
    v8f acc;
#pragma unroll
    for (int v = 0; v < 8; ++v) acc[v] = bv;   // bias broadcast into C
    v16h b0 = load_b_frag(Wt, lane, nb, 0);
    v16h b1 = load_b_frag(Wt, lane, nb, 32);
    acc = __builtin_amdgcn_wmma_f32_16x16x32_f16(false, a0, false, b0,
                                                 (short)0, acc, false, false);
    acc = __builtin_amdgcn_wmma_f32_16x16x32_f16(false, a1, false, b1,
                                                 (short)0, acc, false, false);
#pragma unroll
    for (int v = 0; v < 8; ++v) {
      float x = acc[v];
      if (relu) x = fmaxf(x, 0.0f);
      out[(v + hs * 8) * 64 + nb * 16 + n] = (_Float16)x;
    }
  }
}

__device__ __forceinline__ void layer64_f32in(const float* in, _Float16* out,
                                              const _Float16* Wt, const float* bias,
                                              int lane, bool relu) {
  layer64_core(load_a_frag_f32(in, lane, 0), load_a_frag_f32(in, lane, 32),
               out, Wt, bias, lane, relu);
}

__device__ __forceinline__ void layer64_f16in(const _Float16* in, _Float16* out,
                                              const _Float16* Wt, const float* bias,
                                              int lane, bool relu) {
  layer64_core(load_a_frag(in, lane, 0), load_a_frag(in, lane, 32),
               out, Wt, bias, lane, relu);
}

// acc[4] += in @ W  (f16 input tile, C kept in registers)
__device__ __forceinline__ void accum64(v8f acc[4], const _Float16* in,
                                        const _Float16* Wt, int lane) {
  v16h a0 = load_a_frag(in, lane, 0);
  v16h a1 = load_a_frag(in, lane, 32);
#pragma unroll
  for (int nb = 0; nb < 4; ++nb) {
    v16h b0 = load_b_frag(Wt, lane, nb, 0);
    v16h b1 = load_b_frag(Wt, lane, nb, 32);
    acc[nb] = __builtin_amdgcn_wmma_f32_16x16x32_f16(false, a0, false, b0,
                                                     (short)0, acc[nb], false, false);
    acc[nb] = __builtin_amdgcn_wmma_f32_16x16x32_f16(false, a1, false, b1,
                                                     (short)0, acc[nb], false, false);
  }
}

// Load 64x64 f32 weights (in-major) into LDS transposed (out-major) f16.
template <int TB>
__device__ __forceinline__ void load_wt(_Float16* Wt, const float* W, int tid) {
  for (int i = tid; i < 4096; i += TB) {
    const int n = i >> 6, k = i & 63;
    Wt[n * 64 + k] = (_Float16)W[k * 64 + n];
  }
}

template <int TB>
__device__ __forceinline__ void load_f32v(float* dst, const float* src, int cnt, int tid) {
  for (int i = tid; i < cnt; i += TB) dst[i] = src[i];
}

// Degenerate _eq_msg_stick scale: s = mlp2(q/max(q,1e-12); sw1,sb1,sw2,sb2)
__device__ __forceinline__ float msg_scale(float q, const float* sw1s,
                                           const float* sb1s, const float* sw2s,
                                           float sb2v) {
  const float inv = q / fmaxf(q, 1e-12f);
  float sum = sb2v;
#pragma unroll 8
  for (int j = 0; j < 64; ++j)
    sum += fmaxf(inv * sw1s[j] + sb1s[j], 0.0f) * sw2s[j];
  return sum;
}

// ---------------------------------------------------------------------------
// Kernel 0: initialize d_out = [node_pos | velocity]
// ---------------------------------------------------------------------------
__global__ void gmn_copy(const float* __restrict__ pos, const float* __restrict__ vel,
                         float* __restrict__ outPos, float* __restrict__ outVel, int n3) {
  const int stride = gridDim.x * blockDim.x;
  for (int i = blockIdx.x * blockDim.x + threadIdx.x; i < n3; i += stride) {
    outPos[i] = pos[i];
    outVel[i] = vel[i];
  }
}

// ---------------------------------------------------------------------------
// Kernel 1: isolated nodes.  v = mlp2(h;cv)*vel + frc ; x = pos + v
// ---------------------------------------------------------------------------
__global__ void __launch_bounds__(TBI) gmn_iso(
    const float* __restrict__ feat, const int* __restrict__ iso,
    const float* __restrict__ pos, const float* __restrict__ vel,
    const float* __restrict__ frc,
    const float* __restrict__ cvw1, const float* __restrict__ cvb1,
    const float* __restrict__ cvw2, const float* __restrict__ cvb2,
    float* __restrict__ outPos, float* __restrict__ outVel, int I) {
  __shared__ _Float16 Wt[4096];
  __shared__ float b1s[64], w2s[64];
  __shared__ float    bufF[NWI][1024];     // async-gathered f32 tile
  __shared__ _Float16 bufB[NWI][1024];

  const int tid = threadIdx.x;
  load_wt<TBI>(Wt, cvw1, tid);
  load_f32v<TBI>(b1s, cvb1, 64, tid);
  load_f32v<TBI>(w2s, cvw2, 64, tid);
  __syncthreads();
  const float b2 = cvb2[0];

  const int lane = tid & 31, w = tid >> 5;
  const int nt = (I + 15) >> 4;
  for (int t = blockIdx.x * NWI + w; t < nt; t += gridDim.x * NWI) {
    const int base = t << 4;
    gather_tile_async(bufF[w], feat, iso, base, I, lane);
    wait_async<0>();
    layer64_f32in(bufF[w], bufB[w], Wt, b1s, lane, true);
    lds_fence();
    if (lane < 16) {
      const int s = base + lane;
      if (s < I) {
        float sum = b2;
        const _Float16* row = bufB[w] + lane * 64;
#pragma unroll 8
        for (int k = 0; k < 64; ++k) sum += (float)row[k] * w2s[k];
        const int g = iso[s];
#pragma unroll
        for (int c = 0; c < 3; ++c) {
          const float v = sum * vel[3 * g + c] + frc[3 * g + c];
          outVel[3 * g + c] = v;
          outPos[3 * g + c] = pos[3 * g + c] + v;
        }
      }
    }
    lds_fence();   // bufB reads done before next tile overwrites
  }
}

// ---------------------------------------------------------------------------
// Kernel 2: sticks (reads pos/vel from d_out, i.e. post-isolated update).
// h1 and h2 tile gathers are issued back-to-back as async DMA; in-order
// completion lets s_wait_asynccnt 8 release h1 while h2 is still in flight.
// ---------------------------------------------------------------------------
__global__ void __launch_bounds__(TBS) gmn_stick(
    const float* __restrict__ feat, const int* __restrict__ i1,
    const int* __restrict__ i2, const float* __restrict__ frc,
    const float* __restrict__ cvw1, const float* __restrict__ cvb1,
    const float* __restrict__ cvw2, const float* __restrict__ cvb2,
    const float* __restrict__ avw1, const float* __restrict__ avb1,
    const float* __restrict__ avw2, const float* __restrict__ avb2,
    const float* __restrict__ cw1, const float* __restrict__ cb1,
    const float* __restrict__ cw2, const float* __restrict__ cb2,
    const float* __restrict__ sw1, const float* __restrict__ sb1,
    const float* __restrict__ sw2, const float* __restrict__ sb2,
    float* outPos, float* outVel, int K) {
  __shared__ _Float16 Wc1[4096], Wc2[4096], Wcv[4096], Wav[4096];
  __shared__ float cb1s[64], cb2s[64], cvb1s[64], avb1s[64];
  __shared__ float cvw2s[64], avw2s[64], sw1s[64], sb1s[64], sw2s[64];
  __shared__ float    bufF[NWS][2][1024];  // double-buffered f32 gather tiles
  __shared__ _Float16 bufB[NWS][1024];     // layer outputs (f16)
  __shared__ _Float16 bufH[NWS][1024];     // h_c tile (f16)

  const int tid = threadIdx.x;
  load_wt<TBS>(Wc1, cw1, tid);  load_wt<TBS>(Wc2, cw2, tid);
  load_wt<TBS>(Wcv, cvw1, tid); load_wt<TBS>(Wav, avw1, tid);
  load_f32v<TBS>(cb1s, cb1, 64, tid);   load_f32v<TBS>(cb2s, cb2, 64, tid);
  load_f32v<TBS>(cvb1s, cvb1, 64, tid); load_f32v<TBS>(avb1s, avb1, 64, tid);
  load_f32v<TBS>(cvw2s, cvw2, 64, tid); load_f32v<TBS>(avw2s, avw2, 64, tid);
  load_f32v<TBS>(sw1s, sw1, 64, tid);   load_f32v<TBS>(sb1s, sb1, 64, tid);
  load_f32v<TBS>(sw2s, sw2, 64, tid);
  __syncthreads();
  const float cvb2v = cvb2[0], avb2v = avb2[0], sb2v = sb2[0];

  const int lane = tid & 31, w = tid >> 5;
  const int hs = lane >> 4, n = lane & 15;
  const int nt = (K + 15) >> 4;
  for (int t = blockIdx.x * NWS + w; t < nt; t += gridDim.x * NWS) {
    const int base = t << 4;

    // issue both gathers up front (16 async B128 ops in flight per lane)
    gather_tile_async(bufF[w][0], feat, i1, base, K, lane);   // h1
    gather_tile_async(bufF[w][1], feat, i2, base, K, lane);   // h2

    // h_c accumulators, seeded with 2*cb2 (bias applied once per mlp2)
    v8f hc[4];
#pragma unroll
    for (int nb = 0; nb < 4; ++nb) {
      const float bv = 2.0f * cb2s[nb * 16 + n];
#pragma unroll
      for (int v = 0; v < 8; ++v) hc[nb][v] = bv;
    }

    wait_async<8>();                                          // h1 tile landed
    layer64_f32in(bufF[w][0], bufB[w], Wc1, cb1s, lane, true);// t1
    lds_fence();
    accum64(hc, bufB[w], Wc2, lane);                          // hc += t1@cw2

    wait_async<0>();                                          // h2 tile landed
    layer64_f32in(bufF[w][1], bufB[w], Wc1, cb1s, lane, true);// t2
    lds_fence();
    accum64(hc, bufB[w], Wc2, lane);                          // hc += t2@cw2

    // h_c -> f16 tile (no activation)
#pragma unroll
    for (int nb = 0; nb < 4; ++nb)
#pragma unroll
      for (int v = 0; v < 8; ++v)
        bufH[w][(v + hs * 8) * 64 + nb * 16 + n] = (_Float16)hc[nb][v];
    lds_fence();

    layer64_f16in(bufH[w], bufB[w], Wcv, cvb1s, lane, true);  // relu(h_c@cvw1+b)
    lds_fence();
    float cvs = 0.0f;
    if (lane < 16) {
      float sum = cvb2v;
      const _Float16* row = bufB[w] + lane * 64;
#pragma unroll 8
      for (int k = 0; k < 64; ++k) sum += (float)row[k] * cvw2s[k];
      cvs = sum;
    }
    lds_fence();
    layer64_f16in(bufH[w], bufB[w], Wav, avb1s, lane, true);  // relu(h_c@avw1+b)
    lds_fence();
    float avs = 0.0f;
    if (lane < 16) {
      float sum = avb2v;
      const _Float16* row = bufB[w] + lane * 64;
#pragma unroll 8
      for (int k = 0; k < 64; ++k) sum += (float)row[k] * avw2s[k];
      avs = sum;
    }

    // per-stick rigid-body physics: lane m handles stick base+m
    if (lane < 16) {
      const int s = base + lane;
      if (s < K) {
        const int g1 = i1[s], g2 = i2[s];
        float x1[3], x2[3], V1[3], V2[3], F1[3], F2[3];
#pragma unroll
        for (int c = 0; c < 3; ++c) {
          x1[c] = outPos[3 * g1 + c]; x2[c] = outPos[3 * g2 + c];
          V1[c] = outVel[3 * g1 + c]; V2[c] = outVel[3 * g2 + c];
          F1[c] = frc[3 * g1 + c];    F2[c] = frc[3 * g2 + c];
        }
        const float q1 = F1[0]*F1[0] + F1[1]*F1[1] + F1[2]*F1[2];
        const float q2 = F2[0]*F2[0] + F2[1]*F2[1] + F2[2]*F2[2];
        const float s1 = msg_scale(q1, sw1s, sb1s, sw2s, sb2v);
        const float s2 = msg_scale(q2, sw1s, sb1s, sw2s, sb2v);

        float acc3[3], xc[3], vc[3], r[3], vr[3];
#pragma unroll
        for (int c = 0; c < 3; ++c) {
          acc3[c] = 0.5f * (F1[c] * s1 + F2[c] * s2);
          xc[c] = 0.5f * (x1[c] + x2[c]);
          vc[c] = 0.5f * (V1[c] + V2[c]);
          r[c]  = 0.5f * (x1[c] - x2[c]);
          vr[c] = 0.5f * (V1[c] - V2[c]);
        }
        const float rr2 = r[0]*r[0] + r[1]*r[1] + r[2]*r[2];
        const float inertia = 2.0f * rr2;
        const float df[3] = {F1[0]-F2[0], F1[1]-F2[1], F1[2]-F2[2]};
        // torque = cross(x1-xc,f1) + cross(x2-xc,f2) = cross(r, f1-f2)
        const float tq[3] = {r[1]*df[2]-r[2]*df[1],
                             r[2]*df[0]-r[0]*df[2],
                             r[0]*df[1]-r[1]*df[0]};
        const float rn = sqrtf(rr2);
        const float invh = 1.0f / fmaxf(rn, 1e-12f);
        const float invr = 1.0f / fmaxf(rn, 1e-5f);
        float angv[3];
        angv[0] = (r[1]*vr[2]-r[2]*vr[1]) * invh * invr;
        angv[1] = (r[2]*vr[0]-r[0]*vr[2]) * invh * invr;
        angv[2] = (r[0]*vr[1]-r[1]*vr[0]) * invh * invr;
#pragma unroll
        for (int c = 0; c < 3; ++c) {
          vc[c] = cvs * vc[c] + acc3[c];
          xc[c] += vc[c];
          angv[c] = avs * angv[c] + tq[c] / inertia;
        }
        const float a2 = angv[0]*angv[0] + angv[1]*angv[1] + angv[2]*angv[2];
        const float ang = sqrtf(a2);
        const float dn = 1.0f / fmaxf(ang, 1e-12f);
        const float dx = angv[0]*dn, dy = angv[1]*dn, dz = angv[2]*dn;
        const float cA = cosf(ang), sA = sinf(ang), oc = 1.0f - cA;
        const float rx = (cA+oc*dx*dx)*r[0] + (oc*dx*dy-sA*dz)*r[1] + (oc*dx*dz+sA*dy)*r[2];
        const float ry = (oc*dx*dy+sA*dz)*r[0] + (cA+oc*dy*dy)*r[1] + (oc*dy*dz-sA*dx)*r[2];
        const float rz = (oc*dx*dz-sA*dy)*r[0] + (oc*dy*dz+sA*dx)*r[1] + (cA+oc*dz*dz)*r[2];
        const float wxr[3] = {angv[1]*rz-angv[2]*ry,
                              angv[2]*rx-angv[0]*rz,
                              angv[0]*ry-angv[1]*rx};
        outPos[3*g1+0] = xc[0]+rx; outPos[3*g1+1] = xc[1]+ry; outPos[3*g1+2] = xc[2]+rz;
        outPos[3*g2+0] = xc[0]-rx; outPos[3*g2+1] = xc[1]-ry; outPos[3*g2+2] = xc[2]-rz;
        outVel[3*g1+0] = vc[0]+wxr[0]; outVel[3*g1+1] = vc[1]+wxr[1]; outVel[3*g1+2] = vc[2]+wxr[2];
        outVel[3*g2+0] = vc[0]-wxr[0]; outVel[3*g2+1] = vc[1]-wxr[1]; outVel[3*g2+2] = vc[2]-wxr[2];
      }
    }
    lds_fence();   // bufB/bufH reads done before next tile overwrites
  }
}

// ---------------------------------------------------------------------------
extern "C" void kernel_launch(void* const* d_in, const int* in_sizes, int n_in,
                              void* d_out, int out_size, void* d_ws, size_t ws_size,
                              hipStream_t stream) {
  const float* feat = (const float*)d_in[0];
  const float* pos  = (const float*)d_in[1];
  const float* vel  = (const float*)d_in[2];
  const float* frc  = (const float*)d_in[3];
  const int*   iso  = (const int*)d_in[4];
  const int*   stk  = (const int*)d_in[5];
  const float* cvw1 = (const float*)d_in[6],  *cvb1 = (const float*)d_in[7];
  const float* cvw2 = (const float*)d_in[8],  *cvb2 = (const float*)d_in[9];
  const float* avw1 = (const float*)d_in[10], *avb1 = (const float*)d_in[11];
  const float* avw2 = (const float*)d_in[12], *avb2 = (const float*)d_in[13];
  const float* cw1  = (const float*)d_in[14], *cb1  = (const float*)d_in[15];
  const float* cw2  = (const float*)d_in[16], *cb2  = (const float*)d_in[17];
  const float* sw1  = (const float*)d_in[18], *sb1  = (const float*)d_in[19];
  const float* sw2  = (const float*)d_in[20], *sb2  = (const float*)d_in[21];

  const int N3 = in_sizes[1];         // 3*N
  const int I  = in_sizes[4];
  const int K  = in_sizes[5] / 2;
  const int* i1 = stk;
  const int* i2 = stk + K;

  float* outPos = (float*)d_out;
  float* outVel = outPos + N3;

  gmn_copy<<<2048, 256, 0, stream>>>(pos, vel, outPos, outVel, N3);
  gmn_iso<<<512, TBI, 0, stream>>>(feat, iso, pos, vel, frc,
                                   cvw1, cvb1, cvw2, cvb2, outPos, outVel, I);
  gmn_stick<<<2048, TBS, 0, stream>>>(feat, i1, i2, frc,
                                      cvw1, cvb1, cvw2, cvb2,
                                      avw1, avb1, avw2, avb2,
                                      cw1, cb1, cw2, cb2,
                                      sw1, sb1, sw2, sb2,
                                      outPos, outVel, K);
}
